// MultiModel_34780645163705
// MI455X (gfx1250) — compile-verified, matrix operand order checked by reading, not run on hardware
//
#include <hip/hip_runtime.h>

// ---------------------------------------------------------------------------
// Problem dims: B=32, N=128, D=5000 (padded to KP=5120), H=256, C=4
// Workspace layout (bytes), total ~46.4 MB (x_b aliased by P/S after embed):
//   [0)          x_b   bf16 [4096][5120]   41,943,040   (dead after embed)
//   [0)          P     f32  [4096][1536]   25,165,824   (aliases x_b)
//   [25165824)   S_sh  f32  [32][129][256]  4,227,072
//   [29392896)   S_ru  f32  [32][129][256]  4,227,072
//   [41943040)   Er_b  bf16 [256][5120]     2,621,440
//   [44564480)   Wc_b  bf16 [1536][256]       786,432
//   [45350912)   Usr_b bf16 3x[256][256]      393,216
//   [45744128)   Ucat  bf16 3x[256][512]      786,432
//   [46530560)   xt_b  bf16 [4096][256]     2,097,152
// ---------------------------------------------------------------------------

#define KP 5120

typedef __attribute__((ext_vector_type(16))) __bf16 v16bf;
typedef __attribute__((ext_vector_type(8)))  float  v8f;
typedef __attribute__((ext_vector_type(4)))  int    vint4;   // native int4 vector

#if __has_builtin(__builtin_amdgcn_global_load_async_to_lds_b128) && \
    __has_builtin(__builtin_amdgcn_s_wait_asynccnt)
#define HAVE_ASYNC_LDS 1
typedef __attribute__((address_space(1))) vint4 g_vint4;     // global (__device__) AS
typedef __attribute__((address_space(3))) vint4 l_vint4;     // LDS AS
#else
#define HAVE_ASYNC_LDS 0
#endif

union Frag16 { int4 q[2]; v16bf v; };

__device__ __forceinline__ unsigned short f2b(float f) {
  unsigned int u = __float_as_uint(f);
  u += 0x7FFFu + ((u >> 16) & 1u);      // round-to-nearest-even bf16
  return (unsigned short)(u >> 16);
}

// A/B fragment for v_wmma_f32_16x16x32_bf16 from a row-major [rows][ldk] bf16
// buffer. Lane L supplies row (tile*16 + (L&15)); K-chunks at (L>>4)*8 and
// 16+(L>>4)*8 per CDNA5 ISA 7.12.2 (16-bit A 16x32 layout; B mirrors it).
__device__ __forceinline__ v16bf load_frag(const unsigned short* base, int row,
                                           int ldk, int k, int lane) {
  const unsigned short* p = base + (size_t)row * (size_t)ldk + k + ((lane >> 4) << 3);
  Frag16 f;
  f.q[0] = *(const int4*)(p);
  f.q[1] = *(const int4*)(p + 16);
  return f.v;
}

__device__ __forceinline__ v8f wmma_bf16(v16bf a, v16bf b, v8f c) {
  return __builtin_amdgcn_wmma_f32_16x16x32_bf16(false, a, false, b, (short)0, c,
                                                 false, false);
}

__device__ __forceinline__ float sigmoidf_(float x) {
  return 1.0f / (1.0f + __expf(-x));
}

// ---------------------------------------------------------------------------
// Convert x [4096][5000] f32 -> x_b [4096][5120] bf16 (zero padded tail)
// ---------------------------------------------------------------------------
__global__ void k_convert_x(const float* __restrict__ x, unsigned short* __restrict__ xb) {
  int c = blockIdx.x * 256 + threadIdx.x;      // 0..5119
  int r = blockIdx.y;                          // 0..4095
  if (c < KP) {
    unsigned short v = 0;
    if (c < 5000) v = f2b(x[(size_t)r * 5000 + c]);
    xb[(size_t)r * KP + c] = v;
  }
}

// ---------------------------------------------------------------------------
// Convert Er (padded) + pack W-cat + U + U-cat weight buffers to bf16
// ---------------------------------------------------------------------------
__global__ void k_convert_w(
    const float* __restrict__ Er,
    const float* __restrict__ Wsr_r, const float* __restrict__ Wsr_z, const float* __restrict__ Wsr_h,
    const float* __restrict__ Wr_r,  const float* __restrict__ Wr_z,  const float* __restrict__ Wr_h,
    const float* __restrict__ Usr_r, const float* __restrict__ Usr_z, const float* __restrict__ Usr_h,
    const float* __restrict__ Ur_r,  const float* __restrict__ Usr2r_r,
    const float* __restrict__ Ur_z,  const float* __restrict__ Usr2r_z,
    const float* __restrict__ Ur_h,  const float* __restrict__ Usr2r_h,
    unsigned short* __restrict__ erb, unsigned short* __restrict__ wcb,
    unsigned short* __restrict__ usrb, unsigned short* __restrict__ ucatb)
{
  const int E0 = 256 * KP;              // 1,310,720  Er_b
  const int E1 = E0 + 1536 * 256;       // 1,703,936  Wc_b
  const int E2 = E1 + 3 * 65536;        // 1,900,544  Usr_{r,z,h}
  const int E3 = E2 + 3 * 131072;       // 2,293,760  Ucat_{r,z,h}
  int idx = blockIdx.x * 256 + threadIdx.x;
  if (idx < E0) {
    int r = idx / KP, c = idx % KP;
    erb[idx] = (c < 5000) ? f2b(Er[(size_t)r * 5000 + c]) : (unsigned short)0;
  } else if (idx < E1) {
    int t = idx - E0;
    int r = t / 256, c = t % 256;
    int blk = r >> 8;                   // 0..5 : sr_r, sr_z, sr_h, r_r, r_z, r_h
    const float* W = (blk == 0) ? Wsr_r : (blk == 1) ? Wsr_z : (blk == 2) ? Wsr_h
                   : (blk == 3) ? Wr_r  : (blk == 4) ? Wr_z  : Wr_h;
    wcb[t] = f2b(W[(size_t)(r & 255) * 256 + c]);
  } else if (idx < E2) {
    int t = idx - E1;
    int blk = t / 65536, e = t % 65536; // 0..2 : Usr_r, Usr_z, Usr_h
    const float* U = (blk == 0) ? Usr_r : (blk == 1) ? Usr_z : Usr_h;
    usrb[t] = f2b(U[e]);
  } else if (idx < E3) {
    int t = idx - E2;
    int blk = t / 131072, e = t % 131072;   // 0..2 : cat r, z, h
    int n = e / 512, k = e % 512;           // row n, K = [hrp(256) | h_sh(256)]
    const float* Ua = (blk == 0) ? Ur_r    : (blk == 1) ? Ur_z    : Ur_h;
    const float* Ub = (blk == 0) ? Usr2r_r : (blk == 1) ? Usr2r_z : Usr2r_h;
    float v = (k < 256) ? Ua[(size_t)n * 256 + k] : Ub[(size_t)n * 256 + (k - 256)];
    ucatb[t] = f2b(v);
  }
}

// ---------------------------------------------------------------------------
// Embedding GEMM: xt_b[4096][256] (bf16) = x_b[4096][5120] @ Er_b[256][5120]^T
// grid = 256 (m-tiles), block = 128 (4 waves, each: 1 m-tile x 4 n-tiles).
// A tile (16 rows x 128 K-chunk) staged in LDS via double-buffered async
// global->LDS DMA (ASYNCcnt); B fragments stream from L2-resident Er_b.
// ---------------------------------------------------------------------------
__global__ __launch_bounds__(128) void k_gemm_embed(
    const unsigned short* __restrict__ xb, const unsigned short* __restrict__ erb,
    unsigned short* __restrict__ xtb)
{
  __shared__ unsigned short sA[2][16 * 128];   // 2 x 4 KB double buffer

  const int tid  = threadIdx.x;
  const int lane = tid & 31;
  const int wid  = tid >> 5;           // 0..3
  const int mt   = blockIdx.x;         // 0..255
  const int nt0  = wid * 4;
  const int nl   = lane & 15;
  const int m0   = (lane >> 4) << 3;

  // Issue one stage: copy rows [mt*16, +16) x K-cols [s*128, +128) into sA[s&1].
  // 256 16-byte chunks; each of the 128 threads handles 2.
  auto issue = [&](int s) {
    #pragma unroll
    for (int u = 0; u < 2; ++u) {
      int c   = tid + u * 128;         // 0..255
      int row = c >> 4;                // 0..15
      int col = (c & 15) * 8;          // 0..120
      const unsigned short* g = xb + (size_t)(mt * 16 + row) * KP + s * 128 + col;
      unsigned short* l = &sA[s & 1][row * 128 + col];
#if HAVE_ASYNC_LDS
      __builtin_amdgcn_global_load_async_to_lds_b128((g_vint4*)g, (l_vint4*)l, 0, 0);
#else
      *(int4*)l = *(const int4*)g;
#endif
    }
  };

  v8f acc[4];
  #pragma unroll
  for (int j = 0; j < 4; ++j)
    #pragma unroll
    for (int r = 0; r < 8; ++r) acc[j][r] = 0.0f;

  issue(0);
  for (int s = 0; s < 40; ++s) {
    if (s < 39) {
      issue(s + 1);
#if HAVE_ASYNC_LDS
      __builtin_amdgcn_s_wait_asynccnt(2);   // stage s complete; s+1 in flight
#endif
    } else {
#if HAVE_ASYNC_LDS
      __builtin_amdgcn_s_wait_asynccnt(0);
#endif
    }
    __syncthreads();
    const unsigned short* Abuf = sA[s & 1];
    #pragma unroll
    for (int kk = 0; kk < 128; kk += 32) {
      v16bf a = load_frag(Abuf, nl, 128, kk, lane);
      #pragma unroll
      for (int j = 0; j < 4; ++j) {
        v16bf b = load_frag(erb, (nt0 + j) * 16 + nl, KP, s * 128 + kk, lane);
        acc[j] = wmma_bf16(a, b, acc[j]);
      }
    }
    __syncthreads();
  }

  #pragma unroll
  for (int j = 0; j < 4; ++j)
    #pragma unroll
    for (int r = 0; r < 8; ++r)
      xtb[(size_t)(mt * 16 + m0 + r) * 256 + (nt0 + j) * 16 + nl] = f2b(acc[j][r]);
}

// ---------------------------------------------------------------------------
// W-projection GEMM: P[4096][1536] (f32) = xt_b[4096][256] @ Wc_b[1536][256]^T
// grid = (256 m-tiles, 6 n-supertiles), block = 128 (4 waves x 4 n-tiles)
// ---------------------------------------------------------------------------
__global__ __launch_bounds__(128) void k_gemm_wproj(
    const unsigned short* __restrict__ xtb, const unsigned short* __restrict__ wcb,
    float* __restrict__ P)
{
  const int lane = threadIdx.x & 31;
  const int wid  = threadIdx.x >> 5;
  const int mt   = blockIdx.x;
  const int nl   = lane & 15;
  const int m0   = (lane >> 4) << 3;

  v8f acc[4];
  #pragma unroll
  for (int j = 0; j < 4; ++j)
    #pragma unroll
    for (int r = 0; r < 8; ++r) acc[j][r] = 0.0f;

  const int arow = mt * 16 + nl;
  #pragma unroll
  for (int k = 0; k < 256; k += 32) {
    v16bf a = load_frag(xtb, arow, 256, k, lane);
    #pragma unroll
    for (int j = 0; j < 4; ++j) {
      int nt = blockIdx.y * 16 + wid * 4 + j;     // 0..95
      v16bf b = load_frag(wcb, nt * 16 + nl, 256, k, lane);
      acc[j] = wmma_bf16(a, b, acc[j]);
    }
  }
  #pragma unroll
  for (int j = 0; j < 4; ++j) {
    int nt = blockIdx.y * 16 + wid * 4 + j;
    #pragma unroll
    for (int r = 0; r < 8; ++r)
      P[(size_t)(mt * 16 + m0 + r) * 1536 + nt * 16 + nl] = acc[j][r];
  }
}

// ---------------------------------------------------------------------------
// Sequential tree-GRU scan. One workgroup = 32 wave32 waves on one WGP.
// Per step: gather parent states -> share-GRU (K=256 WMMA) -> rumor-GRU
// (fused [hrp|h_sh] K=512 WMMA). States in global (L2), activations in LDS.
// Dynamic LDS: 160 KB.
// ---------------------------------------------------------------------------
__global__ __launch_bounds__(1024) void k_scan(
    const float* __restrict__ P, const long long* __restrict__ parents,
    const unsigned short* __restrict__ Usr_r, const unsigned short* __restrict__ Usr_z,
    const unsigned short* __restrict__ Usr_h,
    const unsigned short* __restrict__ Ucat_r, const unsigned short* __restrict__ Ucat_z,
    const unsigned short* __restrict__ Ucat_h,
    float* __restrict__ Ssh, float* __restrict__ Sru)
{
  extern __shared__ char smem[];
  unsigned short* sHSP16 = (unsigned short*)(smem);            // [32][256] bf16 hsp
  unsigned short* sRSH16 = (unsigned short*)(smem + 16384);    // [32][256] bf16 hsp*r_sh
  unsigned short* sCATRZ = (unsigned short*)(smem + 32768);    // [32][512] bf16 [hrp | h_sh]
  unsigned short* sCATH  = (unsigned short*)(smem + 65536);    // [32][512] bf16 [hrp*r_ru | h_sh]
  float*          sHSP32 = (float*)(smem + 98304);             // [32][256] f32 hsp
  float*          sHRP32 = (float*)(smem + 131072);            // [32][256] f32 hrp

  const int tid  = threadIdx.x;
  const int lane = tid & 31;
  const int w    = tid >> 5;          // wave 0..31 = output tile coord
  const int mt   = w >> 4;            // 0..1  (batch rows mt*16..+15)
  const int nt   = w & 15;            // 0..15 (hidden cols nt*16..+15)
  const int nl   = lane & 15;
  const int m0   = (lane >> 4) << 3;
  const int arow = mt * 16 + nl;      // A-fragment row (batch row)
  const int brow = nt * 16 + nl;      // B-fragment row (output unit)
  const int hcol = nt * 16 + nl;      // accumulator hidden col

  // zero state row 0 (S[:,0,:] = 0)
  {
    int b = tid >> 5, h0 = (tid & 31) * 8;
    float4 z4 = make_float4(0.f, 0.f, 0.f, 0.f);
    *(float4*)(Ssh + ((size_t)b * 129) * 256 + h0)     = z4;
    *(float4*)(Ssh + ((size_t)b * 129) * 256 + h0 + 4) = z4;
    *(float4*)(Sru + ((size_t)b * 129) * 256 + h0)     = z4;
    *(float4*)(Sru + ((size_t)b * 129) * 256 + h0 + 4) = z4;
  }
  __threadfence();
  __syncthreads();

  for (int i = 0; i < 128; ++i) {
    // Prefetch this step's P-tile segments (6 blocks x 16 rows, 64B each)
    // into cache while the gather + barrier below are in flight.
    #pragma unroll
    for (int q = 0; q < 3; ++q) {
      int e = lane + q * 32;            // 0..95
      int row = e & 15, blk = e >> 4;   // blk 0..5
      __builtin_prefetch(
          &P[((size_t)((mt * 16 + row) * 128 + i)) * 1536 + blk * 256 + nt * 16], 0, 3);
    }

    // ---- P0: gather parent states into LDS (thread t: b=t/32, 8 h-values)
    {
      int b = tid >> 5, h0 = (tid & 31) * 8;
      int p = (int)parents[b * 128 + i];
      const float* psh = Ssh + ((size_t)(b * 129 + p)) * 256 + h0;
      const float* pru = Sru + ((size_t)(b * 129 + p)) * 256 + h0;
      #pragma unroll
      for (int e = 0; e < 8; ++e) {
        float vs = psh[e], vr = pru[e];
        sHSP32[b * 256 + h0 + e] = vs;
        sHSP16[b * 256 + h0 + e] = f2b(vs);
        sHRP32[b * 256 + h0 + e] = vr;
        sCATRZ[b * 512 + h0 + e] = f2b(vr);   // low K-half = hrp
      }
    }
    __syncthreads();

    // ---- P1a: r_sh, z_sh  (each wave: its tile for both gates)
    float zsh[8];
    {
      v8f ar, az;
      #pragma unroll
      for (int r = 0; r < 8; ++r) {
        size_t pr = ((size_t)((mt * 16 + m0 + r) * 128 + i)) * 1536 + hcol;
        ar[r] = P[pr];          // block 0: Wsr_r proj
        az[r] = P[pr + 256];    // block 1: Wsr_z proj
      }
      #pragma unroll
      for (int k = 0; k < 256; k += 32) {
        v16bf a = load_frag(sHSP16, arow, 256, k, lane);
        ar = wmma_bf16(a, load_frag(Usr_r, brow, 256, k, lane), ar);
        az = wmma_bf16(a, load_frag(Usr_z, brow, 256, k, lane), az);
      }
      #pragma unroll
      for (int r = 0; r < 8; ++r) {
        int b = mt * 16 + m0 + r;
        float rsh = sigmoidf_(ar[r]);
        zsh[r]    = sigmoidf_(az[r]);
        sRSH16[b * 256 + hcol] = f2b(sHSP32[b * 256 + hcol] * rsh);
      }
    }
    __syncthreads();

    // ---- P1b: h_sh
    {
      v8f ah;
      #pragma unroll
      for (int r = 0; r < 8; ++r)
        ah[r] = P[((size_t)((mt * 16 + m0 + r) * 128 + i)) * 1536 + 2 * 256 + hcol];
      #pragma unroll
      for (int k = 0; k < 256; k += 32) {
        v16bf a = load_frag(sRSH16, arow, 256, k, lane);
        ah = wmma_bf16(a, load_frag(Usr_h, brow, 256, k, lane), ah);
      }
      #pragma unroll
      for (int r = 0; r < 8; ++r) {
        int b = mt * 16 + m0 + r;
        float hsp = sHSP32[b * 256 + hcol];
        float hsh = (1.0f - zsh[r]) * hsp + zsh[r] * tanhf(ah[r]);
        Ssh[((size_t)(b * 129 + i + 1)) * 256 + hcol] = hsh;
        unsigned short u = f2b(hsh);
        sCATRZ[b * 512 + 256 + hcol] = u;     // high K-half = h_sh
        sCATH [b * 512 + 256 + hcol] = u;
      }
    }
    __syncthreads();

    // ---- P2a: r_ru, z_ru  (fused K=512: [hrp|h_sh] @ [Ur_*|Usr2r_*]^T)
    float zru[8];
    {
      v8f ar, az;
      #pragma unroll
      for (int r = 0; r < 8; ++r) {
        size_t pr = ((size_t)((mt * 16 + m0 + r) * 128 + i)) * 1536 + hcol;
        ar[r] = P[pr + 3 * 256];    // Wr_r proj
        az[r] = P[pr + 4 * 256];    // Wr_z proj
      }
      #pragma unroll
      for (int k = 0; k < 512; k += 32) {
        v16bf a = load_frag(sCATRZ, arow, 512, k, lane);
        ar = wmma_bf16(a, load_frag(Ucat_r, brow, 512, k, lane), ar);
        az = wmma_bf16(a, load_frag(Ucat_z, brow, 512, k, lane), az);
      }
      #pragma unroll
      for (int r = 0; r < 8; ++r) {
        int b = mt * 16 + m0 + r;
        float rru = sigmoidf_(ar[r]);
        zru[r]    = sigmoidf_(az[r]);
        sCATH[b * 512 + hcol] = f2b(sHRP32[b * 256 + hcol] * rru);
      }
    }
    __syncthreads();

    // ---- P2b: h_ru
    {
      v8f ah;
      #pragma unroll
      for (int r = 0; r < 8; ++r)
        ah[r] = P[((size_t)((mt * 16 + m0 + r) * 128 + i)) * 1536 + 5 * 256 + hcol];
      #pragma unroll
      for (int k = 0; k < 512; k += 32) {
        v16bf a = load_frag(sCATH, arow, 512, k, lane);
        ah = wmma_bf16(a, load_frag(Ucat_h, brow, 512, k, lane), ah);
      }
      #pragma unroll
      for (int r = 0; r < 8; ++r) {
        int b = mt * 16 + m0 + r;
        float hrp = sHRP32[b * 256 + hcol];
        float hru = (1.0f - zru[r]) * hrp + zru[r] * tanhf(ah[r]);
        Sru[((size_t)(b * 129 + i + 1)) * 256 + hcol] = hru;
      }
    }
    __threadfence();   // make S_sh/S_ru stores visible before next P0 gather
    __syncthreads();
  }
}

// ---------------------------------------------------------------------------
// Leaf-masked max pool over nodes + logits (Vr,br) + softmax. 1 block per b.
// ---------------------------------------------------------------------------
__global__ __launch_bounds__(256) void k_pool(
    const float* __restrict__ Sru, const unsigned char* __restrict__ leaf,
    const float* __restrict__ Vr, const float* __restrict__ br,
    float* __restrict__ out)
{
  __shared__ float hmax[256];
  int b = blockIdx.x, h = threadIdx.x;
  float m = -3.402823466e38f;
  for (int i = 0; i < 128; ++i)
    if (leaf[b * 128 + i])
      m = fmaxf(m, Sru[((size_t)(b * 129 + i + 1)) * 256 + h]);
  hmax[h] = m;
  __syncthreads();
  if (h == 0) {
    float lg[4];
    #pragma unroll
    for (int c = 0; c < 4; ++c) {
      float s = br[c];
      for (int k = 0; k < 256; ++k) s += hmax[k] * Vr[c * 256 + k];
      lg[c] = s;
    }
    float mx = fmaxf(fmaxf(lg[0], lg[1]), fmaxf(lg[2], lg[3]));
    float e[4], den = 0.0f;
    #pragma unroll
    for (int c = 0; c < 4; ++c) { e[c] = __expf(lg[c] - mx); den += e[c]; }
    #pragma unroll
    for (int c = 0; c < 4; ++c) out[b * 4 + c] = e[c] / den;
  }
}

// ---------------------------------------------------------------------------
extern "C" void kernel_launch(void* const* d_in, const int* in_sizes, int n_in,
                              void* d_out, int out_size, void* d_ws, size_t ws_size,
                              hipStream_t stream) {
  (void)in_sizes; (void)n_in; (void)out_size; (void)ws_size;
  const float* x       = (const float*)d_in[0];
  const float* Er      = (const float*)d_in[1];
  const float* Wsr_r   = (const float*)d_in[2];
  const float* Usr_r   = (const float*)d_in[3];
  const float* Wsr_z   = (const float*)d_in[4];
  const float* Usr_z   = (const float*)d_in[5];
  const float* Wsr_h   = (const float*)d_in[6];
  const float* Usr_h   = (const float*)d_in[7];
  const float* Wr_r    = (const float*)d_in[8];
  const float* Ur_r    = (const float*)d_in[9];
  const float* Usr2r_r = (const float*)d_in[10];
  const float* Wr_z    = (const float*)d_in[11];
  const float* Ur_z    = (const float*)d_in[12];
  const float* Usr2r_z = (const float*)d_in[13];
  const float* Wr_h    = (const float*)d_in[14];
  const float* Ur_h    = (const float*)d_in[15];
  const float* Usr2r_h = (const float*)d_in[16];
  const float* Vr      = (const float*)d_in[17];
  const float* br      = (const float*)d_in[18];
  const long long* parents   = (const long long*)d_in[19];
  const unsigned char* leaf  = (const unsigned char*)d_in[20];
  float* out = (float*)d_out;

  char* ws = (char*)d_ws;
  unsigned short* xb    = (unsigned short*)(ws + 0);          // dead after embed
  float*          Pbuf  = (float*)(ws + 0);                   // aliases xb
  float*          Ssh   = (float*)(ws + 25165824);
  float*          Sru   = (float*)(ws + 29392896);
  unsigned short* erb   = (unsigned short*)(ws + 41943040);
  unsigned short* wcb   = (unsigned short*)(ws + 44564480);
  unsigned short* usrb  = (unsigned short*)(ws + 45350912);
  unsigned short* ucatb = (unsigned short*)(ws + 45744128);
  unsigned short* xtb   = (unsigned short*)(ws + 46530560);

  k_convert_x<<<dim3(20, 4096), 256, 0, stream>>>(x, xb);
  k_convert_w<<<8960, 256, 0, stream>>>(Er,
      Wsr_r, Wsr_z, Wsr_h, Wr_r, Wr_z, Wr_h,
      Usr_r, Usr_z, Usr_h,
      Ur_r, Usr2r_r, Ur_z, Usr2r_z, Ur_h, Usr2r_h,
      erb, wcb, usrb, ucatb);
  k_gemm_embed<<<256, 128, 0, stream>>>(xb, erb, xtb);
  k_gemm_wproj<<<dim3(256, 6), 128, 0, stream>>>(xtb, wcb, Pbuf);
  k_scan<<<1, 1024, 163840, stream>>>(Pbuf, parents,
      usrb, usrb + 65536, usrb + 131072,
      ucatb, ucatb + 131072, ucatb + 262144,
      Ssh, Sru);
  k_pool<<<32, 256, 0, stream>>>(Sru, leaf, Vr, br, out);
}
